// GroupedQueryAttention_48206712930694
// MI455X (gfx1250) — matrix operand order repaired and checked by reading, not compiled, not run on hardware
//
#include <hip/hip_runtime.h>

// ---------------------------------------------------------------------------
// CDNA5 (gfx1250) GQA: bf16 WMMA GEMMs + flash attention, fp32 accumulate.
// Hot loops: double-buffered global_load_async_to_lds_b128 (ASYNCcnt) + WMMA.
// ---------------------------------------------------------------------------

typedef __attribute__((ext_vector_type(16))) __bf16 v16bf;
typedef __attribute__((ext_vector_type(8)))  float  v8f;

#define WMMA_BF16(a, b, c) \
  __builtin_amdgcn_wmma_f32_16x16x32_bf16(false, (a), false, (b), (short)0, (c), false, false)

__device__ __forceinline__ unsigned short f2bf(float f) {
  union { float f; unsigned u; } x; x.f = f;
  unsigned r = x.u + 0x7fffu + ((x.u >> 16) & 1u);   // round-to-nearest-even
  return (unsigned short)(r >> 16);
}

union FragU { v16bf v; uint4 q[2]; };

// A-matrix fragment (16x32 bf16). ISA layout: lane r=lane&15 is row M=r,
// kh=lane>>4 selects K runs {kh*8..+8} and {16+kh*8..+8} (two 16B loads).
__device__ __forceinline__ v16bf frag_a(const unsigned short* base, int rowStride, int kbase) {
  const int lane = threadIdx.x & 31;
  const int r = lane & 15, kh = lane >> 4;
  const unsigned short* p = base + (size_t)r * rowStride + kbase + kh * 8;
  FragU f;
  f.q[0] = *(const uint4*)(p);
  f.q[1] = *(const uint4*)(p + 16);
  return f.v;
}

// B-matrix fragment (32x16 bf16) read from an [N][K] (K-contiguous) tile.
// ISA layout: lane c=lane&15 is column N=c, kh=lane>>4 selects K runs {kh*8..+8} and {16+kh*8..+8}.
__device__ __forceinline__ v16bf frag_b(const unsigned short* base, int rowStride, int cbase, int kbase) {
  const int lane = threadIdx.x & 31;
  const int c = lane & 15, kh = lane >> 4;
  const unsigned short* p = base + (size_t)(cbase + c) * rowStride + kbase + kh * 8;
  FragU f;
  f.q[0] = *(const uint4*)(p);
  f.q[1] = *(const uint4*)(p + 16);
  return f.v;
}

// Async global->LDS copy, 16 bytes per lane. Tracked by ASYNCcnt.
// GV mode: LDS address comes from the vdst VGPR; low 32 bits of a flat LDS pointer
// are the wave-relative LDS byte address (aperture truncation).
__device__ __forceinline__ void async_b128(void* lds, const void* gptr) {
  unsigned la = (unsigned)(__SIZE_TYPE__)lds;
  unsigned long long ga = (unsigned long long)(__SIZE_TYPE__)gptr;
  asm volatile("global_load_async_to_lds_b128 %0, %1, off"
               :: "v"(la), "v"(ga) : "memory");
}
__device__ __forceinline__ void wait_async() {
  asm volatile("s_wait_asynccnt 0x0" ::: "memory");
}

// ---------------------------------------------------------------------------
__global__ void gqa_cvt_bf16(const float* __restrict__ s, unsigned short* __restrict__ d, size_t n) {
  size_t i = (size_t)blockIdx.x * blockDim.x + threadIdx.x;
  const size_t stride = (size_t)gridDim.x * blockDim.x;
  for (; i < n; i += stride) d[i] = f2bf(s[i]);
}

// W[K][N] f32 -> Wt[N][K] bf16. N = 1<<nShift (power of two: shift/mask only).
__global__ void gqa_cvt_tr_bf16(const float* __restrict__ w, unsigned short* __restrict__ wt,
                                int K, int nShift) {
  const unsigned nMask = (1u << nShift) - 1u;
  size_t i = (size_t)blockIdx.x * blockDim.x + threadIdx.x;
  const size_t stride = (size_t)gridDim.x * blockDim.x;
  const size_t total = (size_t)K << nShift;
  for (; i < total; i += stride) {
    const int k = (int)(i >> nShift), n = (int)(i & nMask);
    wt[(size_t)n * K + k] = f2bf(w[i]);
  }
}

// ---------------------------------------------------------------------------
// C[M,N] = A[M,K](bf16) @ W[K,N] + bias, with W pre-transposed: Wt[N][K].
// 256 thr = 8 waves (4x2), block tile 128x128, K-step 32, double-buffered LDS.
// Each wave: 32x64 -> 2 A-frags x 4 B-frags = 8 WMMAs per K-step.
// MODE: 0 = f32 store, 1 = bf16 store, 2 = bf16 store transposed for V
//       (out index [(b*8+h)*64+d][s], b=row>>11, s=row&2047, h=col>>6, d=col&63)
// ---------------------------------------------------------------------------
template <int MODE>
__global__ __launch_bounds__(256) void gqa_gemm_bf16(
    const unsigned short* __restrict__ A, const unsigned short* __restrict__ Wt,
    const float* __restrict__ bias, float* __restrict__ Cf, unsigned short* __restrict__ Cb,
    int N, int K)
{
  __shared__ unsigned short As[2][128 * 40];   // [128 rows][32 K] pad->40
  __shared__ unsigned short Bs[2][128 * 40];   // [128 cols][32 K] pad->40

  const int n0 = blockIdx.x * 128, m0 = blockIdx.y * 128;
  const int tid = threadIdx.x, wid = tid >> 5, lane = tid & 31;
  const int wM = wid >> 1, wN = wid & 1;

  v8f acc[2][4];
  #pragma unroll
  for (int i = 0; i < 2; ++i)
    #pragma unroll
    for (int t = 0; t < 4; ++t)
      #pragma unroll
      for (int j = 0; j < 8; ++j) acc[i][t][j] = 0.f;

  const int sRow = (tid * 16) >> 5;   // 0..127
  const int sCol = (tid * 16) & 31;   // 0 or 16
  const unsigned short* ga = A  + (size_t)(m0 + sRow) * K + sCol;
  const unsigned short* gb = Wt + (size_t)(n0 + sRow) * K + sCol;
  const int sOff = sRow * 40 + sCol;

  auto issue = [&](int buf, int kb) {
    async_b128(&As[buf][sOff],     ga + kb);
    async_b128(&As[buf][sOff + 8], ga + kb + 8);
    async_b128(&Bs[buf][sOff],     gb + kb);
    async_b128(&Bs[buf][sOff + 8], gb + kb + 8);
  };

  const int niter = K >> 5;
  issue(0, 0);
  for (int it = 0; it < niter; ++it) {
    wait_async();          // tile `it` landed in LDS
    __syncthreads();       // everyone sees it; prev compute done reading buf it^1
    if (it + 1 < niter) issue((it + 1) & 1, (it + 1) << 5);   // DMA overlaps WMMAs

    const unsigned short* as = As[it & 1];
    const unsigned short* bs = Bs[it & 1];
    const v16bf a0 = frag_a(as + (wM * 32) * 40, 40, 0);
    const v16bf a1 = frag_a(as + (wM * 32 + 16) * 40, 40, 0);
    #pragma unroll
    for (int t = 0; t < 4; ++t) {
      const v16bf b = frag_b(bs, 40, wN * 64 + t * 16, 0);
      acc[0][t] = WMMA_BF16(a0, b, acc[0][t]);
      acc[1][t] = WMMA_BF16(a1, b, acc[1][t]);
    }
  }

  // C layout: lane holds col N=lane&15 (+tile), rows (lane>>4)*8 + j.
  const int hh = lane >> 4, cc = lane & 15;
  #pragma unroll
  for (int t = 0; t < 4; ++t) {
    const int col = n0 + wN * 64 + t * 16 + cc;
    const float bv = bias[col];
    #pragma unroll
    for (int i = 0; i < 2; ++i) {
      #pragma unroll
      for (int j = 0; j < 8; ++j) {
        const int row = m0 + wM * 32 + i * 16 + hh * 8 + j;
        const float v = acc[i][t][j] + bv;
        if constexpr (MODE == 0) {
          Cf[(size_t)row * N + col] = v;
        } else if constexpr (MODE == 1) {
          Cb[(size_t)row * N + col] = f2bf(v);
        } else {
          const int bb = row >> 11, s = row & 2047;
          const int h = col >> 6, d = col & 63;
          Cb[(((size_t)(bb * 8 + h) * 64 + d) << 11) + s] = f2bf(v);
        }
      }
    }
  }
}

// ---------------------------------------------------------------------------
// Flash attention, causal, GQA (G=4). One block = 128 queries of one q-head.
// 8 waves x 16 rows; each wave owns all 64 key columns per key block, so the
// online softmax row-reduction is intra-wave (shfl within 16-lane halves).
// K tile used K-contiguous as-is; V pre-transposed VT[(b*8+h)*64+d][s].
// K/V tiles double-buffered via async global->LDS b128.
// ---------------------------------------------------------------------------
__global__ __launch_bounds__(256) void gqa_attn_bf16(
    const unsigned short* __restrict__ Q, const unsigned short* __restrict__ Kg,
    const unsigned short* __restrict__ VT, unsigned short* __restrict__ O)
{
  const int S = 2048, DQ = 2048, DKV = 512, HD = 64;
  const int qb = blockIdx.x, hq = blockIdx.y, b = blockIdx.z;
  const int h = hq >> 2;   // G = 4

  __shared__ unsigned short Ks[2][64 * 72];   // [key][d]
  __shared__ unsigned short Vs[2][64 * 72];   // [d][key]
  __shared__ unsigned short Ps[128 * 72];     // P round-trip C-layout -> A-layout

  const int tid = threadIdx.x, wid = tid >> 5, lane = tid & 31;
  const int r = lane & 15, hh = lane >> 4;

  // Q fragments straight from global (A-layout rows are contiguous in HD).
  const unsigned short* qbase = Q + ((size_t)b * S + qb * 128 + wid * 16) * DQ + hq * HD;
  const v16bf aQ0 = frag_a(qbase, DQ, 0);
  const v16bf aQ1 = frag_a(qbase, DQ, 32);

  float m[8], l[8];
  v8f o[4];
  #pragma unroll
  for (int j = 0; j < 8; ++j) { m[j] = -1e30f; l[j] = 0.f; }
  #pragma unroll
  for (int t = 0; t < 4; ++t)
    #pragma unroll
    for (int j = 0; j < 8; ++j) o[t][j] = 0.f;

  const int ldRow = (tid * 16) >> 6;   // 0..63
  const int ldCol = (tid * 16) & 63;   // 0,16,32,48
  const int ldOff = ldRow * 72 + ldCol;
  const float scale = 0.125f;          // 1/sqrt(64)
  const int nkb = qb * 2 + 2;          // causal: only key blocks <= query tile

  const unsigned short* krow = Kg + ((size_t)b * S + ldRow) * DKV + h * HD + ldCol;
  const unsigned short* vrow = VT + (((size_t)(b * 8 + h) * 64 + ldRow) << 11) + ldCol;

  auto issueKV = [&](int buf, int kb) {
    const unsigned short* kp = krow + (size_t)kb * 64 * DKV;
    const unsigned short* vp = vrow + kb * 64;
    async_b128(&Ks[buf][ldOff],     kp);
    async_b128(&Ks[buf][ldOff + 8], kp + 8);
    async_b128(&Vs[buf][ldOff],     vp);
    async_b128(&Vs[buf][ldOff + 8], vp + 8);
  };

  issueKV(0, 0);
  for (int kb = 0; kb < nkb; ++kb) {
    wait_async();
    __syncthreads();
    if (kb + 1 < nkb) issueKV((kb + 1) & 1, kb + 1);   // DMA overlaps compute

    const unsigned short* ks = Ks[kb & 1];
    const unsigned short* vs = Vs[kb & 1];

    // S = Q * K^T  (4 col-tiles x 2 K-steps = 8 WMMAs per wave)
    v8f s[4];
    #pragma unroll
    for (int t = 0; t < 4; ++t) {
      #pragma unroll
      for (int j = 0; j < 8; ++j) s[t][j] = 0.f;
      s[t] = WMMA_BF16(aQ0, frag_b(ks, 72, t * 16, 0),  s[t]);
      s[t] = WMMA_BF16(aQ1, frag_b(ks, 72, t * 16, 32), s[t]);
    }

    // scale + causal mask (C-layout: col = t*16 + (lane&15), row = hh*8 + j)
    const int q0 = qb * 128 + wid * 16 + hh * 8;
    #pragma unroll
    for (int t = 0; t < 4; ++t) {
      const int kidx = kb * 64 + t * 16 + r;
      #pragma unroll
      for (int j = 0; j < 8; ++j) {
        float v = s[t][j] * scale;
        if (kidx > q0 + j) v = -1e30f;
        s[t][j] = v;
      }
    }

    // row max over 64 cols: across 4 tiles in-register, then across the
    // 16 lanes of this half-wave (xor 1/2/4/8 never crosses the halves).
    float corr[8];
    #pragma unroll
    for (int j = 0; j < 8; ++j) {
      float v = fmaxf(fmaxf(s[0][j], s[1][j]), fmaxf(s[2][j], s[3][j]));
      v = fmaxf(v, __shfl_xor(v, 1));
      v = fmaxf(v, __shfl_xor(v, 2));
      v = fmaxf(v, __shfl_xor(v, 4));
      v = fmaxf(v, __shfl_xor(v, 8));
      const float mn = fmaxf(m[j], v);
      corr[j] = __expf(m[j] - mn);
      m[j] = mn;
    }

    float rs[8];
    #pragma unroll
    for (int j = 0; j < 8; ++j) rs[j] = 0.f;
    #pragma unroll
    for (int t = 0; t < 4; ++t)
      #pragma unroll
      for (int j = 0; j < 8; ++j) {
        const float p = __expf(s[t][j] - m[j]);
        s[t][j] = p;
        rs[j] += p;
      }
    #pragma unroll
    for (int j = 0; j < 8; ++j) {
      float v = rs[j];
      v += __shfl_xor(v, 1);
      v += __shfl_xor(v, 2);
      v += __shfl_xor(v, 4);
      v += __shfl_xor(v, 8);
      l[j] = l[j] * corr[j] + v;
    }
    #pragma unroll
    for (int t = 0; t < 4; ++t)
      #pragma unroll
      for (int j = 0; j < 8; ++j) o[t][j] *= corr[j];

    // P: C-layout registers -> LDS (bf16), re-read in A-layout.
    #pragma unroll
    for (int t = 0; t < 4; ++t)
      #pragma unroll
      for (int j = 0; j < 8; ++j)
        Ps[(wid * 16 + hh * 8 + j) * 72 + t * 16 + r] = f2bf(s[t][j]);

    __syncthreads();

    const v16bf aP0 = frag_a(Ps + wid * 16 * 72, 72, 0);
    const v16bf aP1 = frag_a(Ps + wid * 16 * 72, 72, 32);
    #pragma unroll
    for (int t = 0; t < 4; ++t) {
      o[t] = WMMA_BF16(aP0, frag_b(vs, 72, t * 16, 0),  o[t]);
      o[t] = WMMA_BF16(aP1, frag_b(vs, 72, t * 16, 32), o[t]);
    }
  }

  #pragma unroll
  for (int j = 0; j < 8; ++j) l[j] = 1.f / l[j];
  #pragma unroll
  for (int t = 0; t < 4; ++t)
    #pragma unroll
    for (int j = 0; j < 8; ++j) {
      const int row = qb * 128 + wid * 16 + hh * 8 + j;
      O[((size_t)b * S + row) * DQ + hq * HD + t * 16 + r] = f2bf(o[t][j] * l[j]);
    }
}

// ---------------------------------------------------------------------------
extern "C" void kernel_launch(void* const* d_in, const int* in_sizes, int n_in,
                              void* d_out, int out_size, void* d_ws, size_t ws_size,
                              hipStream_t stream)
{
  (void)in_sizes; (void)n_in; (void)out_size; (void)ws_size;
  const float* x  = (const float*)d_in[0];
  const float* Wq = (const float*)d_in[1];
  const float* bq = (const float*)d_in[2];
  const float* Wk = (const float*)d_in[3];
  const float* bk = (const float*)d_in[4];
  const float* Wv = (const float*)d_in[5];
  const float* bv = (const float*)d_in[6];
  const float* Wo = (const float*)d_in[7];
  const float* bo = (const float*)d_in[8];
  float* out = (float*)d_out;

  const int B = 2, S = 2048, D = 2048, HK = 8, HD = 64;
  const int M = B * S, DKV = HK * HD;   // 4096, 512

  char* ws = (char*)d_ws;
  size_t off = 0;
  auto bump = [&](size_t elems) {
    unsigned short* p = (unsigned short*)(ws + off);
    off += elems * sizeof(unsigned short);
    return p;
  };
  unsigned short* xb  = bump((size_t)M * D);     // x (bf16)
  unsigned short* Wqt = bump((size_t)D * D);     // weights transposed [N][K] bf16
  unsigned short* Wkt = bump((size_t)D * DKV);
  unsigned short* Wvt = bump((size_t)D * DKV);
  unsigned short* Wot = bump((size_t)D * D);
  unsigned short* Qb  = bump((size_t)M * D);     // Q bf16 [tok][hq*64+d]
  unsigned short* Kb  = bump((size_t)M * DKV);   // K bf16 [tok][h*64+d]
  unsigned short* VTb = bump((size_t)M * DKV);   // V bf16 transposed [(b*8+h)*64+d][s]
  unsigned short* Ab  = bump((size_t)M * D);     // attention out (bf16)

  gqa_cvt_bf16<<<1024, 256, 0, stream>>>(x, xb, (size_t)M * D);
  gqa_cvt_tr_bf16<<<1024, 256, 0, stream>>>(Wq, Wqt, D, 11);   // N=2048
  gqa_cvt_tr_bf16<<<1024, 256, 0, stream>>>(Wk, Wkt, D, 9);    // N=512
  gqa_cvt_tr_bf16<<<1024, 256, 0, stream>>>(Wv, Wvt, D, 9);
  gqa_cvt_tr_bf16<<<1024, 256, 0, stream>>>(Wo, Wot, D, 11);

  gqa_gemm_bf16<1><<<dim3(D / 128,   M / 128), 256, 0, stream>>>(xb, Wqt, bq, nullptr, Qb,  D,   D);
  gqa_gemm_bf16<1><<<dim3(DKV / 128, M / 128), 256, 0, stream>>>(xb, Wkt, bk, nullptr, Kb,  DKV, D);
  gqa_gemm_bf16<2><<<dim3(DKV / 128, M / 128), 256, 0, stream>>>(xb, Wvt, bv, nullptr, VTb, DKV, D);

  gqa_attn_bf16<<<dim3(S / 128, 32, B), 256, 0, stream>>>(Qb, Kb, VTb, Ab);

  gqa_gemm_bf16<0><<<dim3(D / 128, M / 128), 256, 0, stream>>>(Ab, Wot, bo, out, nullptr, D, D);
}
